// SwinTransformerBlock_28286654611671
// MI455X (gfx1250) — matrix operand order repaired and checked by reading, not compiled, not run on hardware
//
#include <hip/hip_runtime.h>

// ---------------- problem constants ----------------
#define LTOT   86016      // D*H*W tokens
#define NWIN   1344       // number of 4x4x4 windows
#define CDIM   96
#define NH     6
#define HIDF   384
#define DD     32
#define HH     48
#define WWID   56

typedef __attribute__((ext_vector_type(16))) _Float16 v16h;
typedef __attribute__((ext_vector_type(8)))  float    v8f;
typedef unsigned int u32x4 __attribute__((ext_vector_type(4)));
typedef int i32x8 __attribute__((ext_vector_type(8)));
typedef int i32x4 __attribute__((ext_vector_type(4)));

#if defined(__has_builtin)
#  if __has_builtin(__builtin_amdgcn_tensor_load_to_lds) && __has_builtin(__builtin_amdgcn_s_wait_tensorcnt)
#    define USE_TDM 1
#  else
#    define USE_TDM 0
#  endif
#else
#  define USE_TDM 0
#endif

__device__ __forceinline__ v8f wmma16(v16h a, v16h b, v8f c) {
  return __builtin_amdgcn_wmma_f32_16x16x32_f16(false, a, false, b, (short)0, c, false, false);
}

// ---------------- Tensor Data Mover staging ----------------
#if USE_TDM
__device__ __forceinline__ void tdm_issue_2d(void* lds_dst, const void* gsrc,
                                             unsigned rows, unsigned cols,
                                             unsigned long stride_elems) {
  const unsigned long ga = (unsigned long)gsrc;
  u32x4 g0;
  g0[0] = 1u;                                          // count=1, user descriptor
  g0[1] = (unsigned)(unsigned long)lds_dst;            // LDS byte address (addr[31:0])
  g0[2] = (unsigned)(ga & 0xffffffffu);                // global_addr[31:0]
  g0[3] = (unsigned)((ga >> 32) & 0x01ffffffu) | 0x80000000u;  // addr[56:32] | type=2
  i32x8 g1;
  g1[0] = (int)(1u << 16);                             // data_size=1 -> 2-byte elems
  g1[1] = (int)((cols & 0xffffu) << 16);               // tensor_dim0[15:0]
  g1[2] = (int)(((cols >> 16) & 0xffffu) | ((rows & 0xffffu) << 16)); // dim0 hi | dim1 lo
  g1[3] = (int)(((rows >> 16) & 0xffffu) | ((cols & 0xffffu) << 16)); // dim1 hi | tile_dim0
  g1[4] = (int)(rows & 0xffffu);                       // tile_dim1 (tile_dim2=0)
  g1[5] = (int)(stride_elems & 0xffffffffu);           // tensor_dim0_stride[31:0]
  g1[6] = (int)((stride_elems >> 32) & 0xffffu);       // stride0[47:32] (stride1=0)
  g1[7] = 0;
  const i32x4 z4 = {0, 0, 0, 0};
  const i32x8 z8 = {0, 0, 0, 0, 0, 0, 0, 0};
  __builtin_amdgcn_tensor_load_to_lds(g0, g1, z4, z4, z8, 0);
}
#endif

__device__ __forceinline__ void stage2d(_Float16* dst, const _Float16* src,
                                        int rows, int cols, long srcStride,
                                        int tid, int wave) {
#if USE_TDM
  (void)tid;
  if (wave == 0)
    tdm_issue_2d(dst, src, (unsigned)rows, (unsigned)cols, (unsigned long)srcStride);
#else
  (void)wave;
  for (int e = tid; e < rows * cols; e += 256)
    dst[e] = src[(size_t)(e / cols) * (size_t)srcStride + (e % cols)];
#endif
}
__device__ __forceinline__ void stage_commit(int wave) {
#if USE_TDM
  if (wave == 0) __builtin_amdgcn_s_wait_tensorcnt(0);
#else
  (void)wave;
#endif
}

// ---------------- WMMA fragment loaders (contiguous-k layouts) ----------------
// A 16x32 f16 from row-major LDS: per-lane halves are 16 contiguous k
// (k in [k0, k0+16), k0 = 16*(lane/16)) -> 2x ds_load_b128.
__device__ __forceinline__ v16h frag_a(const _Float16* p, int lda) {
  const int lane = threadIdx.x & 31;
  const int m = lane & 15;
  const int k0 = (lane >> 4) * 16;
  union { u32x4 q[2]; v16h v; } r;
  const _Float16* base = p + m * lda + k0;
  r.q[0] = *(const u32x4*)(base);        // halves k = k0 .. k0+7
  r.q[1] = *(const u32x4*)(base + 8);    // halves k = k0+8 .. k0+15
  return r.v;
}
// A 16x16 tile (lda=16), K padded to 32 with zeros
__device__ __forceinline__ v16h frag_aq_k16(const _Float16* p) {
  const int lane = threadIdx.x & 31;
  const int m = lane & 15;
  union { u32x4 q[2]; v16h v; } r;
  r.q[0] = (u32x4){0, 0, 0, 0};                // lanes 16-31: k >= 16 -> zero pad
  r.q[1] = (u32x4){0, 0, 0, 0};
  if ((lane >> 4) == 0) {                      // lanes 0-15: k = 0..15
    r.q[0] = *(const u32x4*)(p + m * 16);
    r.q[1] = *(const u32x4*)(p + m * 16 + 8);
  }
  return r.v;
}
// B 32x16 from TRANSPOSED [n][k] LDS layout: lane's 16 halves contiguous
__device__ __forceinline__ v16h frag_bt(const _Float16* p, int ldn) {
  const int lane = threadIdx.x & 31;
  const int n = lane & 15;
  const int k0 = (lane >> 4) * 16;
  union { u32x4 q[2]; v16h v; } r;
  const _Float16* base = p + n * ldn + k0;
  r.q[0] = *(const u32x4*)(base);
  r.q[1] = *(const u32x4*)(base + 8);
  return r.v;
}
// B from a 16-token x 16-dim row-major K tile (= transposed layout), K padded
__device__ __forceinline__ v16h frag_bt_k16(const _Float16* p) {
  const int lane = threadIdx.x & 31;
  const int n = lane & 15;
  v16h z;
#pragma unroll
  for (int j = 0; j < 16; ++j) z[j] = (_Float16)0.0f;
  if ((lane >> 4) == 0) {                 // lanes 0-15: k = 0..15 (valid)
    union { u32x4 q[2]; v16h v; } r;
    const _Float16* base = p + n * 16;
    r.q[0] = *(const u32x4*)(base);
    r.q[1] = *(const u32x4*)(base + 8);
    z = r.v;
  }
  return z;                               // lanes 16-31: k = 16..31 -> zero
}
// generic strided B (K x N row-major), used where transpose is unavailable
__device__ __forceinline__ v16h frag_b(const _Float16* p, int ldb) {
  const int lane = threadIdx.x & 31;
  const int n = lane & 15;
  const int k0 = (lane >> 4) * 16;
  v16h b;
#pragma unroll
  for (int j = 0; j < 16; ++j) b[j] = p[(k0 + j) * ldb + n];
  return b;
}

// ---------------- wave32 reductions ----------------
__device__ __forceinline__ float hsum16(float v) {
  v += __shfl_xor(v, 1, 32); v += __shfl_xor(v, 2, 32);
  v += __shfl_xor(v, 4, 32); v += __shfl_xor(v, 8, 32);
  return v;
}
__device__ __forceinline__ float hmax16(float v) {
  v = fmaxf(v, __shfl_xor(v, 1, 32)); v = fmaxf(v, __shfl_xor(v, 2, 32));
  v = fmaxf(v, __shfl_xor(v, 4, 32)); v = fmaxf(v, __shfl_xor(v, 8, 32));
  return v;
}
__device__ __forceinline__ float wsum32(float v) {
  v += __shfl_xor(v, 16, 32);
  return hsum16(v);
}

// ---------------- weight repack (fp32 -> fp16, keep row-major) ----------------
__global__ void __launch_bounds__(256) k_convP(const float* __restrict__ src,
                                               _Float16* __restrict__ dst, int n) {
  for (int e = blockIdx.x * blockDim.x + threadIdx.x; e < n; e += gridDim.x * blockDim.x)
    dst[e] = (_Float16)src[e];
}

// ---------------- fused LN1 + QKV GEMM, window-gathered ----------------
// weights row-major (288,96) fp16; B panels are contiguous 96-row slices
__global__ void __launch_bounds__(256) k_qkv(
    const float* __restrict__ xin, const float* __restrict__ yin,
    const float* __restrict__ g0, const float* __restrict__ b0,
    const float* __restrict__ g1, const float* __restrict__ b1,
    const _Float16* __restrict__ wq0, const _Float16* __restrict__ wq1,
    const float* __restrict__ qb0, const float* __restrict__ qb1,
    _Float16* __restrict__ o0, _Float16* __restrict__ o1)
{
  const int w = blockIdx.x, br = blockIdx.y;
  const float* xp = br ? yin : xin;
  const float* lng = br ? g1 : g0;
  const float* lnb = br ? b1 : b0;
  const _Float16* wq = br ? wq1 : wq0;
  const float* qb = br ? qb1 : qb0;
  _Float16* outp = br ? o1 : o0;

  __shared__ _Float16 sA[64][104];
  __shared__ _Float16 sB[96][96];   // [n=o_local][k=c] (transposed-B = weight rows)
  const int tid = threadIdx.x, lane = tid & 31, wave = tid >> 5;
  const int wd = w / 168, wrem = w % 168, wh = wrem / 14, ww = wrem % 14;

  for (int rr = 0; rr < 8; ++rr) {
    const int n = wave * 8 + rr;
    const int id = n >> 4, ih = (n >> 2) & 3, iw = n & 3;
    const size_t t = ((size_t)(wd * 4 + id) * HH + (wh * 4 + ih)) * WWID + (ww * 4 + iw);
    const float* row = xp + t * CDIM;
    float v0 = row[lane], v1 = row[lane + 32], v2 = row[lane + 64];
    float s  = wsum32(v0 + v1 + v2);
    float sq = wsum32(v0 * v0 + v1 * v1 + v2 * v2);
    float mean = s * (1.0f / 96.0f);
    float var  = sq * (1.0f / 96.0f) - mean * mean;
    float rstd = rsqrtf(var + 1e-5f);
    sA[n][lane]      = (_Float16)((v0 - mean) * rstd * lng[lane]      + lnb[lane]);
    sA[n][lane + 32] = (_Float16)((v1 - mean) * rstd * lng[lane + 32] + lnb[lane + 32]);
    sA[n][lane + 64] = (_Float16)((v2 - mean) * rstd * lng[lane + 64] + lnb[lane + 64]);
  }
  __syncthreads();

  for (int ch = 0; ch < 3; ++ch) {          // q / k / v output chunk (96 rows)
    stage2d(&sB[0][0], wq + (size_t)ch * 96 * 96, 1, 9216, 9216, tid, wave);
    stage_commit(wave);
    __syncthreads();
    for (int i = 0; i < 3; ++i) {
      const int tile = wave + 8 * i;        // 24 tiles = 4M x 6N
      const int mt = tile / 6, nt = tile % 6;
      v8f acc = {0.f,0.f,0.f,0.f,0.f,0.f,0.f,0.f};
#pragma unroll
      for (int ks = 0; ks < 3; ++ks) {
        v16h a = frag_a(&sA[mt * 16][ks * 32], 104);
        v16h b = frag_bt(&sB[nt * 16][ks * 32], 96);
        acc = wmma16(a, b, acc);
      }
      const int nn = lane & 15, g = lane >> 4;
      const int ocol = ch * 96 + nt * 16 + nn;
      const float bias = qb[ocol];
#pragma unroll
      for (int r = 0; r < 8; ++r) {
        const int row = mt * 16 + r + 8 * g;
        outp[((size_t)w * 64 + row) * 288 + ocol] = (_Float16)(acc[r] + bias);
      }
    }
    __syncthreads();
  }
}

// ---------------- cross-window attention (both out-branches per block) ----------------
__global__ void __launch_bounds__(256) k_attn(
    const _Float16* __restrict__ qkx, const _Float16* __restrict__ qky,
    const float* __restrict__ rpb,
    _Float16* __restrict__ ax, _Float16* __restrict__ ay)
{
  const int w = blockIdx.x, head = blockIdx.y;
  __shared__ _Float16 sT[6][64][16];   // qx,ky,vy,qy,kx,vx
  __shared__ float    sBias[64][64];
  __shared__ _Float16 sP[8][16][64];

  const int tid = threadIdx.x, lane = tid & 31, wave = tid >> 5;
  const int sel = wave >> 2;   // 0: x-out, 1: y-out
  const int mt  = wave & 3;

  for (int tt = 0; tt < 6; ++tt) {
    const _Float16* sp = (tt == 0 || tt == 4 || tt == 5) ? qkx : qky;
    const int cbase = (tt == 0 || tt == 3) ? head * 16
                    : ((tt == 1 || tt == 4) ? 96 + head * 16 : 192 + head * 16);
    stage2d(&sT[tt][0][0], sp + (size_t)w * 64 * 288 + cbase, 64, 16, 288, tid, wave);
  }
  for (int e = tid; e < 64 * 64; e += 256) {
    const int m = e >> 6, n = e & 63;
    const int dm = (m >> 4) - (n >> 4) + 3;
    const int hm = ((m >> 2) & 3) - ((n >> 2) & 3) + 3;
    const int wm = (m & 3) - (n & 3) + 3;
    sBias[m][n] = rpb[(dm * 49 + hm * 7 + wm) * NH + head];
  }
  stage_commit(wave);
  __syncthreads();

  const _Float16* sQ = &sT[sel * 3 + 0][0][0];
  const _Float16* sK = &sT[sel * 3 + 1][0][0];
  const _Float16* sV = &sT[sel * 3 + 2][0][0];

  const int g = lane >> 4, nn = lane & 15;
  float S[4][8];
#pragma unroll
  for (int nt = 0; nt < 4; ++nt) {
    v8f acc = {0.f,0.f,0.f,0.f,0.f,0.f,0.f,0.f};
    v16h a = frag_aq_k16(sQ + mt * 16 * 16);
    v16h b = frag_bt_k16(sK + nt * 16 * 16);
    acc = wmma16(a, b, acc);
#pragma unroll
    for (int r = 0; r < 8; ++r) {
      const int m = mt * 16 + r + 8 * g;
      const int n = nt * 16 + nn;
      S[nt][r] = acc[r] * 0.25f + sBias[m][n];   // SCALE = 16^-0.5
    }
  }
#pragma unroll
  for (int r = 0; r < 8; ++r) {
    float mx = fmaxf(fmaxf(S[0][r], S[1][r]), fmaxf(S[2][r], S[3][r]));
    mx = hmax16(mx);
    float sum = 0.0f;
#pragma unroll
    for (int nt = 0; nt < 4; ++nt) { S[nt][r] = __expf(S[nt][r] - mx); sum += S[nt][r]; }
    sum = hsum16(sum);
    const float inv = 1.0f / sum;
#pragma unroll
    for (int nt = 0; nt < 4; ++nt) S[nt][r] *= inv;
  }
#pragma unroll
  for (int nt = 0; nt < 4; ++nt)
#pragma unroll
    for (int r = 0; r < 8; ++r)
      sP[wave][r + 8 * g][nt * 16 + nn] = (_Float16)S[nt][r];

  v8f oacc = {0.f,0.f,0.f,0.f,0.f,0.f,0.f,0.f};
#pragma unroll
  for (int ks = 0; ks < 2; ++ks) {
    v16h a = frag_a(&sP[wave][0][ks * 32], 64);
    v16h b = frag_b(sV + ks * 32 * 16, 16);   // V tile: transpose unavailable
    oacc = wmma16(a, b, oacc);
  }
  _Float16* outp = sel ? ay : ax;
#pragma unroll
  for (int r = 0; r < 8; ++r) {
    const int row = mt * 16 + r + 8 * g;
    outp[((size_t)w * 64 + row) * CDIM + head * 16 + nn] = (_Float16)oacc[r];
  }
}

// ---------------- proj GEMM + window-reverse + residual ----------------
__global__ void __launch_bounds__(256) k_proj(
    const _Float16* __restrict__ a0, const _Float16* __restrict__ a1,
    const _Float16* __restrict__ w0, const _Float16* __restrict__ w1,
    const float* __restrict__ pb0, const float* __restrict__ pb1,
    const float* __restrict__ xin, const float* __restrict__ yin,
    float* __restrict__ r0, float* __restrict__ r1)
{
  const int w = blockIdx.x, br = blockIdx.y;
  const _Float16* A = br ? a1 : a0;
  const _Float16* W = br ? w1 : w0;
  const float* pb = br ? pb1 : pb0;
  const float* xi = br ? yin : xin;
  float* out = br ? r1 : r0;

  __shared__ _Float16 sA[64][96];
  __shared__ _Float16 sB[96][96];   // [n=o][k=c]
  const int tid = threadIdx.x, lane = tid & 31, wave = tid >> 5;
  stage2d(&sA[0][0], A + (size_t)w * 64 * 96, 1, 6144, 6144, tid, wave);
  stage2d(&sB[0][0], W, 1, 9216, 9216, tid, wave);
  stage_commit(wave);
  __syncthreads();

  const int wd = w / 168, wrem = w % 168, wh = wrem / 14, ww = wrem % 14;
  for (int i = 0; i < 3; ++i) {
    const int tile = wave + 8 * i;
    const int mt = tile / 6, nt = tile % 6;
    v8f acc = {0.f,0.f,0.f,0.f,0.f,0.f,0.f,0.f};
#pragma unroll
    for (int ks = 0; ks < 3; ++ks) {
      v16h a = frag_a(&sA[mt * 16][ks * 32], 96);
      v16h b = frag_bt(&sB[nt * 16][ks * 32], 96);
      acc = wmma16(a, b, acc);
    }
    const int nn = lane & 15, g = lane >> 4;
    const int o = nt * 16 + nn;
    const float bias = pb[o];
#pragma unroll
    for (int r = 0; r < 8; ++r) {
      const int row = mt * 16 + r + 8 * g;
      const int id = row >> 4, ih = (row >> 2) & 3, iw = row & 3;
      const size_t t = ((size_t)(wd * 4 + id) * HH + (wh * 4 + ih)) * WWID + (ww * 4 + iw);
      out[t * CDIM + o] = xi[t * CDIM + o] + acc[r] + bias;
    }
  }
}

// ---------------- fused LN2 + L1 GEMM + exact GELU (+ transposed conv input) ----------------
__global__ void __launch_bounds__(256) k_mlp1(
    const float* __restrict__ x0, const float* __restrict__ x1p,
    const float* __restrict__ g0, const float* __restrict__ b0,
    const float* __restrict__ g1, const float* __restrict__ b1,
    const _Float16* __restrict__ w0, const _Float16* __restrict__ w1,
    const float* __restrict__ bb0, const float* __restrict__ bb1,
    _Float16* __restrict__ h0, _Float16* __restrict__ h1,
    _Float16* __restrict__ c0, _Float16* __restrict__ c1)
{
  const int blk = blockIdx.x, br = blockIdx.y;
  const float* xin = br ? x1p : x0;
  const float* lng = br ? g1 : g0;
  const float* lnb = br ? b1 : b0;
  const _Float16* w1w = br ? w1 : w0;
  const float* l1b = br ? bb1 : bb0;
  _Float16* hid = br ? h1 : h0;
  _Float16* cb  = br ? c1 : c0;

  __shared__ _Float16 sA[64][104];
  __shared__ _Float16 sB[96][96];   // [n=o_local][k=c]
  const int tid = threadIdx.x, lane = tid & 31, wave = tid >> 5;
  const int t0 = blk * 64;

  for (int rr = 0; rr < 8; ++rr) {
    const int n = wave * 8 + rr;
    const size_t t = (size_t)t0 + n;
    const float* row = xin + t * CDIM;
    float v0 = row[lane], v1 = row[lane + 32], v2 = row[lane + 64];
    float s  = wsum32(v0 + v1 + v2);
    float sq = wsum32(v0 * v0 + v1 * v1 + v2 * v2);
    float mean = s * (1.0f / 96.0f);
    float var  = sq * (1.0f / 96.0f) - mean * mean;
    float rstd = rsqrtf(var + 1e-5f);
    float e0 = (v0 - mean) * rstd * lng[lane]      + lnb[lane];
    float e1 = (v1 - mean) * rstd * lng[lane + 32] + lnb[lane + 32];
    float e2 = (v2 - mean) * rstd * lng[lane + 64] + lnb[lane + 64];
    sA[n][lane] = (_Float16)e0; sA[n][lane + 32] = (_Float16)e1; sA[n][lane + 64] = (_Float16)e2;
    cb[(size_t)lane        * LTOT + t] = (_Float16)e0;
    cb[(size_t)(lane + 32) * LTOT + t] = (_Float16)e1;
    cb[(size_t)(lane + 64) * LTOT + t] = (_Float16)e2;
  }
  __syncthreads();

  for (int ch = 0; ch < 4; ++ch) {        // 384 = 4 chunks of 96 output rows
    stage2d(&sB[0][0], w1w + (size_t)ch * 96 * 96, 1, 9216, 9216, tid, wave);
    stage_commit(wave);
    __syncthreads();
    for (int i = 0; i < 3; ++i) {
      const int tile = wave + 8 * i;
      const int mt = tile / 6, nt = tile % 6;
      v8f acc = {0.f,0.f,0.f,0.f,0.f,0.f,0.f,0.f};
#pragma unroll
      for (int ks = 0; ks < 3; ++ks) {
        v16h a = frag_a(&sA[mt * 16][ks * 32], 104);
        v16h b = frag_bt(&sB[nt * 16][ks * 32], 96);
        acc = wmma16(a, b, acc);
      }
      const int nn = lane & 15, g = lane >> 4;
      const int o = ch * 96 + nt * 16 + nn;
      const float bias = l1b[o];
#pragma unroll
      for (int r = 0; r < 8; ++r) {
        const int row = mt * 16 + r + 8 * g;
        float v = acc[r] + bias;
        v = 0.5f * v * (1.0f + erff(v * 0.70710678118654752440f));
        hid[(size_t)(t0 + row) * HIDF + o] = (_Float16)v;
      }
    }
    __syncthreads();
  }
}

// ---------------- L2 GEMM (384 -> 96) ----------------
__global__ void __launch_bounds__(256) k_mlp2(
    const _Float16* __restrict__ h0, const _Float16* __restrict__ h1,
    const _Float16* __restrict__ w0, const _Float16* __restrict__ w1,
    const float* __restrict__ b0, const float* __restrict__ b1,
    float* __restrict__ l0, float* __restrict__ l1p)
{
  const int blk = blockIdx.x, br = blockIdx.y;
  const _Float16* hid = br ? h1 : h0;
  const _Float16* w2 = br ? w1 : w0;
  const float* l2b = br ? b1 : b0;
  float* lin = br ? l1p : l0;

  __shared__ _Float16 sA[64][96];
  __shared__ _Float16 sB[96][96];   // [n=o][k=h_local]
  const int tid = threadIdx.x, lane = tid & 31, wave = tid >> 5;
  const int t0 = blk * 64;

  v8f acc[3];
#pragma unroll
  for (int i = 0; i < 3; ++i) acc[i] = (v8f){0.f,0.f,0.f,0.f,0.f,0.f,0.f,0.f};

  for (int kc = 0; kc < 4; ++kc) {
    stage2d(&sA[0][0], hid + (size_t)t0 * HIDF + kc * 96, 64, 96, HIDF, tid, wave);
    stage2d(&sB[0][0], w2 + kc * 96, 96, 96, HIDF, tid, wave);  // rows o, cols h-chunk
    stage_commit(wave);
    __syncthreads();
    for (int i = 0; i < 3; ++i) {
      const int tile = wave + 8 * i;
      const int mt = tile / 6, nt = tile % 6;
#pragma unroll
      for (int ks = 0; ks < 3; ++ks) {
        v16h a = frag_a(&sA[mt * 16][ks * 32], 96);
        v16h b = frag_bt(&sB[nt * 16][ks * 32], 96);
        acc[i] = wmma16(a, b, acc[i]);
      }
    }
    __syncthreads();
  }
  const int nn = lane & 15, g = lane >> 4;
  for (int i = 0; i < 3; ++i) {
    const int tile = wave + 8 * i;
    const int mt = tile / 6, nt = tile % 6;
    const int o = nt * 16 + nn;
    const float bias = l2b[o];
#pragma unroll
    for (int r = 0; r < 8; ++r) {
      const int row = mt * 16 + r + 8 * g;
      lin[(size_t)(t0 + row) * CDIM + o] = acc[i][r] + bias;
    }
  }
}

// ---------------- depthwise 3x3x3 conv, SAME padding, (C,D,H,W) layout ----------------
__global__ void __launch_bounds__(256) k_dw(
    const _Float16* __restrict__ i0, const _Float16* __restrict__ i1,
    const float* __restrict__ w0, const float* __restrict__ w1,
    const float* __restrict__ b0, const float* __restrict__ b1,
    _Float16* __restrict__ o0, _Float16* __restrict__ o1)
{
  const int br = blockIdx.y;
  const _Float16* in = br ? i1 : i0;
  const float* wt = br ? w1 : w0;
  const float* bs = br ? b1 : b0;
  _Float16* out = br ? o1 : o0;
  const int HWp = HH * WWID;
  for (size_t idx = (size_t)blockIdx.x * blockDim.x + threadIdx.x;
       idx < (size_t)CDIM * LTOT; idx += (size_t)gridDim.x * blockDim.x) {
    const int c = (int)(idx / LTOT);
    const int s = (int)(idx % LTOT);
    const int z = s / HWp, r2 = s % HWp, yy = r2 / WWID, xx = r2 % WWID;
    float sum = bs[c];
    const float* wc = wt + c * 27;
#pragma unroll
    for (int dz = -1; dz <= 1; ++dz) {
      const int zz = z + dz; if (zz < 0 || zz >= DD) continue;
#pragma unroll
      for (int dy = -1; dy <= 1; ++dy) {
        const int yv = yy + dy; if (yv < 0 || yv >= HH) continue;
#pragma unroll
        for (int dx = -1; dx <= 1; ++dx) {
          const int xv = xx + dx; if (xv < 0 || xv >= WWID) continue;
          sum += (float)in[((size_t)c * DD + zz) * HWp + (size_t)yv * WWID + xv] *
                 wc[(dz + 1) * 9 + (dy + 1) * 3 + (dx + 1)];
        }
      }
    }
    out[idx] = (_Float16)sum;
  }
}

// ---------------- pointwise (1x1x1) conv as WMMA channel GEMM ----------------
__global__ void __launch_bounds__(256) k_pw(
    const _Float16* __restrict__ i0, const _Float16* __restrict__ i1,
    const _Float16* __restrict__ w0, const _Float16* __restrict__ w1,
    const float* __restrict__ b0, const float* __restrict__ b1,
    _Float16* __restrict__ o0, _Float16* __restrict__ o1)
{
  const int blk = blockIdx.x, br = blockIdx.y;
  const _Float16* in = br ? i1 : i0;
  const _Float16* W = br ? w1 : w0;
  const float* bs = br ? b1 : b0;
  _Float16* out = br ? o1 : o0;

  __shared__ _Float16 sW[96][96];   // A: [co][ci]
  __shared__ _Float16 sI[96][64];   // B: [ci][t] (K x N; transpose unavailable)
  const int tid = threadIdx.x, lane = tid & 31, wave = tid >> 5;
  const int t0 = blk * 64;
  stage2d(&sW[0][0], W, 1, 9216, 9216, tid, wave);
  stage2d(&sI[0][0], in + t0, 96, 64, LTOT, tid, wave);
  stage_commit(wave);
  __syncthreads();

  for (int i = 0; i < 3; ++i) {
    const int tile = wave + 8 * i;      // 24 tiles = 6(co) x 4(t)
    const int mt = tile / 4, nt = tile % 4;
    v8f acc = {0.f,0.f,0.f,0.f,0.f,0.f,0.f,0.f};
#pragma unroll
    for (int ks = 0; ks < 3; ++ks) {
      v16h a = frag_a(&sW[mt * 16][ks * 32], 96);
      v16h b = frag_b(&sI[ks * 32][nt * 16], 64);
      acc = wmma16(a, b, acc);
    }
    const int nn = lane & 15, g = lane >> 4;
    const int t = t0 + nt * 16 + nn;
#pragma unroll
    for (int r = 0; r < 8; ++r) {
      const int co = mt * 16 + r + 8 * g;
      out[(size_t)co * LTOT + t] = (_Float16)(acc[r] + bs[co]);
    }
  }
}

// ---------------- final: out = x1 + lin + raw-view(conv) ----------------
__global__ void __launch_bounds__(256) k_final(
    const float* __restrict__ x0, const float* __restrict__ x1p,
    const float* __restrict__ l0, const float* __restrict__ l1p,
    const _Float16* __restrict__ c0, const _Float16* __restrict__ c1,
    float* __restrict__ out)
{
  const int br = blockIdx.y;
  const float* xv = br ? x1p : x0;
  const float* lv = br ? l1p : l0;
  const _Float16* cv = br ? c1 : c0;
  float* o = out + (size_t)br * LTOT * CDIM;
  for (size_t idx = (size_t)blockIdx.x * blockDim.x + threadIdx.x;
       idx < (size_t)LTOT * CDIM; idx += (size_t)gridDim.x * blockDim.x)
    o[idx] = xv[idx] + lv[idx] + (float)cv[idx];
}

// ======================= host =======================
extern "C" void kernel_launch(void* const* d_in, const int* in_sizes, int n_in,
                              void* d_out, int out_size, void* d_ws, size_t ws_size,
                              hipStream_t stream)
{
  if (n_in < 45) return;
  const float* in[64];
  for (int i = 0; i < n_in && i < 64; ++i) in[i] = (const float*)d_in[i];

  const float *X, *Y, *RPB;
  const float *N1G[2], *N1B[2], *N2G[2], *N2B[2];
  const float *QW[2], *QB[2], *PW_[2], *PB[2];
  const float *L1W[2], *L1B[2], *L2W[2], *L2B[2];
  const float *DW1W[2], *DW1B[2], *PW1W[2], *PW1B[2];
  const float *DW2W[2], *DW2B[2], *PW2W[2], *PW2B[2];

  if (in_sizes[0] == LTOT * CDIM) {
    X = in[0]; Y = in[1]; RPB = in[4];
    for (int b = 0; b < 2; ++b) {
      const int o = 5 + 20 * b;
      N1G[b]=in[o+0]; N1B[b]=in[o+1]; N2G[b]=in[o+2]; N2B[b]=in[o+3];
      QW[b]=in[o+4];  QB[b]=in[o+5];  PW_[b]=in[o+6]; PB[b]=in[o+7];
      L1W[b]=in[o+8]; L1B[b]=in[o+9]; L2W[b]=in[o+10]; L2B[b]=in[o+11];
      DW1W[b]=in[o+12]; DW1B[b]=in[o+13]; PW1W[b]=in[o+14]; PW1B[b]=in[o+15];
      DW2W[b]=in[o+16]; DW2B[b]=in[o+17]; PW2W[b]=in[o+18]; PW2B[b]=in[o+19];
    }
  } else {
    for (int b = 0; b < 2; ++b) {
      const int o = 2 + 12 * b;
      DW1B[b]=in[o+0]; DW1W[b]=in[o+1]; DW2B[b]=in[o+2]; DW2W[b]=in[o+3];
      L1B[b]=in[o+4];  L1W[b]=in[o+5];  L2B[b]=in[o+6];  L2W[b]=in[o+7];
      PW1B[b]=in[o+8]; PW1W[b]=in[o+9]; PW2B[b]=in[o+10]; PW2W[b]=in[o+11];
    }
    N1B[0]=in[26]; N1G[0]=in[27]; N1B[1]=in[28]; N1G[1]=in[29];
    N2B[0]=in[30]; N2G[0]=in[31]; N2B[1]=in[32]; N2G[1]=in[33];
    PB[0]=in[34];  PW_[0]=in[35]; PB[1]=in[36];  PW_[1]=in[37];
    QB[0]=in[38];  QW[0]=in[39];  QB[1]=in[40];  QW[1]=in[41];
    RPB = in[42]; X = in[43]; Y = in[44];
  }

  char* base = (char*)d_ws;
  size_t off = 0;
  auto take = [&](size_t bytes) -> void* {
    void* p = base + off;
    off += (bytes + 255) & ~(size_t)255;
    return p;
  };
  _Float16 *wq[2], *wp[2], *w1[2], *w2[2], *wpw1[2], *wpw2[2];
  _Float16 *qkvb[2], *attnb[2], *hid[2], *cb0[2], *cb1[2];
  float *x1[2], *lin[2];
  for (int b = 0; b < 2; ++b) wq[b]   = (_Float16*)take((size_t)288 * 96 * 2);
  for (int b = 0; b < 2; ++b) wp[b]   = (_Float16*)take((size_t)96 * 96 * 2);
  for (int b = 0; b < 2; ++b) w1[b]   = (_Float16*)take((size_t)HIDF * 96 * 2);
  for (int b = 0; b < 2; ++b) w2[b]   = (_Float16*)take((size_t)96 * HIDF * 2);
  for (int b = 0; b < 2; ++b) wpw1[b] = (_Float16*)take((size_t)96 * 96 * 2);
  for (int b = 0; b < 2; ++b) wpw2[b] = (_Float16*)take((size_t)96 * 96 * 2);
  for (int b = 0; b < 2; ++b) qkvb[b] = (_Float16*)take((size_t)NWIN * 64 * 288 * 2);
  for (int b = 0; b < 2; ++b) attnb[b]= (_Float16*)take((size_t)NWIN * 64 * 96 * 2);
  for (int b = 0; b < 2; ++b) hid[b]  = (_Float16*)take((size_t)LTOT * HIDF * 2);
  for (int b = 0; b < 2; ++b) cb0[b]  = (_Float16*)take((size_t)CDIM * LTOT * 2);
  for (int b = 0; b < 2; ++b) cb1[b]  = (_Float16*)take((size_t)CDIM * LTOT * 2);
  for (int b = 0; b < 2; ++b) x1[b]   = (float*)take((size_t)LTOT * CDIM * 4);
  for (int b = 0; b < 2; ++b) lin[b]  = (float*)take((size_t)LTOT * CDIM * 4);
  if (off > ws_size) return;

  // all weights stay row-major; fp32 -> fp16 plain converts
  for (int b = 0; b < 2; ++b) {
    k_convP<<<dim3(108), 256, 0, stream>>>(QW[b],   wq[b],   288 * 96);
    k_convP<<<dim3(36),  256, 0, stream>>>(PW_[b],  wp[b],   96 * 96);
    k_convP<<<dim3(144), 256, 0, stream>>>(L1W[b],  w1[b],   HIDF * 96);
    k_convP<<<dim3(144), 256, 0, stream>>>(L2W[b],  w2[b],   96 * HIDF);
    k_convP<<<dim3(36),  256, 0, stream>>>(PW1W[b], wpw1[b], 96 * 96);
    k_convP<<<dim3(36),  256, 0, stream>>>(PW2W[b], wpw2[b], 96 * 96);
  }

  k_qkv<<<dim3(NWIN, 2), 256, 0, stream>>>(X, Y, N1G[0], N1B[0], N1G[1], N1B[1],
                                           wq[0], wq[1], QB[0], QB[1], qkvb[0], qkvb[1]);
  k_attn<<<dim3(NWIN, NH), 256, 0, stream>>>(qkvb[0], qkvb[1], RPB, attnb[0], attnb[1]);
  k_proj<<<dim3(NWIN, 2), 256, 0, stream>>>(attnb[0], attnb[1], wp[0], wp[1],
                                            PB[0], PB[1], X, Y, x1[0], x1[1]);
  k_mlp1<<<dim3(NWIN, 2), 256, 0, stream>>>(x1[0], x1[1], N2G[0], N2B[0], N2G[1], N2B[1],
                                            w1[0], w1[1], L1B[0], L1B[1],
                                            hid[0], hid[1], cb0[0], cb0[1]);
  k_mlp2<<<dim3(NWIN, 2), 256, 0, stream>>>(hid[0], hid[1], w2[0], w2[1],
                                            L2B[0], L2B[1], lin[0], lin[1]);
  k_dw<<<dim3(8192, 2), 256, 0, stream>>>(cb0[0], cb0[1], DW1W[0], DW1W[1],
                                          DW1B[0], DW1B[1], cb1[0], cb1[1]);
  k_pw<<<dim3(NWIN, 2), 256, 0, stream>>>(cb1[0], cb1[1], wpw1[0], wpw1[1],
                                          PW1B[0], PW1B[1], cb0[0], cb0[1]);
  k_dw<<<dim3(8192, 2), 256, 0, stream>>>(cb0[0], cb0[1], DW2W[0], DW2W[1],
                                          DW2B[0], DW2B[1], cb1[0], cb1[1]);
  k_pw<<<dim3(NWIN, 2), 256, 0, stream>>>(cb1[0], cb1[1], wpw2[0], wpw2[1],
                                          PW2B[0], PW2B[1], cb0[0], cb0[1]);
  k_final<<<dim3(4096, 2), 256, 0, stream>>>(x1[0], x1[1], lin[0], lin[1],
                                             cb0[0], cb0[1], (float*)d_out);
}
